// MultiHeadAttention_11656541241637
// MI455X (gfx1250) — matrix operand order repaired and checked by reading, not compiled, not run on hardware
//
#include <hip/hip_runtime.h>
#include <hip/hip_bf16.h>
#include <stdint.h>

typedef __attribute__((ext_vector_type(16))) _Float16 v16h;
typedef __attribute__((ext_vector_type(8)))  float    v8f;
typedef __attribute__((ext_vector_type(4)))  int      v4i;

constexpr int Bc = 4, Sc = 2048, Dc = 1024, Hc = 16, HDc = 64;
constexpr int MTOT = Bc * Sc;  // 8192 rows

// ---- CDNA5 async global->LDS path (guarded; falls back to uint4 copies) ----
#if defined(__has_builtin)
#if __has_builtin(__builtin_amdgcn_global_load_async_to_lds_b128) && \
    __has_builtin(__builtin_amdgcn_s_wait_asynccnt)
#define USE_ASYNC_LDS 1
#endif
#endif

// builtin signature (from compiler diagnostic): (AS1 v4i*, AS3 v4i*, imm, imm)
#define AS1V4I(p) ((__attribute__((address_space(1))) v4i*)(v4i*)(void*)(p))
#define AS3V4I(p) ((__attribute__((address_space(3))) v4i*)(v4i*)(void*)(p))

// copy 16 bytes global->LDS (per lane), async if available
__device__ __forceinline__ void copy16_g2l(_Float16* lds_dst, const _Float16* g_src) {
#ifdef USE_ASYNC_LDS
  __builtin_amdgcn_global_load_async_to_lds_b128(AS1V4I(g_src), AS3V4I(lds_dst), 0, 0);
#else
  *reinterpret_cast<uint4*>(lds_dst) = *reinterpret_cast<const uint4*>(g_src);
#endif
}

__device__ __forceinline__ void async_fence() {
#ifdef USE_ASYNC_LDS
  __builtin_amdgcn_s_wait_asynccnt(0);
#endif
}

// ---------------------------------------------------------------------------
// Projection GEMM: Y16[b,h,s,hd] = f16( X[b*s, :] @ W + bias )
// Block tile 128x128, K-step 32, 8 waves, one 16x128 wave-tile each.
// ---------------------------------------------------------------------------
__global__ __launch_bounds__(256)
void proj_gemm_f16(const float* __restrict__ X, const float* __restrict__ W,
                   const float* __restrict__ bias, _Float16* __restrict__ Y) {
  __shared__ _Float16 ldsA[128 * 32];   // [m][k]
  __shared__ _Float16 ldsBT[128 * 32];  // [n][k] (transposed for contiguous B-frags)
  const int tid = threadIdx.x;
  const int lane = tid & 31, wave = tid >> 5;
  const int l16 = lane & 15, hi = lane >> 4;
  const int row0 = blockIdx.x * 128;
  const int col0 = blockIdx.y * 128;

  v8f acc[8] = {};
  for (int k0 = 0; k0 < Dc; k0 += 32) {
    if (k0 + 32 < Dc) {  // prefetch next K-step tiles into L2 (global_prefetch_b8)
      __builtin_prefetch(X + (size_t)(row0 + (tid >> 1)) * Dc + k0 + 32 + (tid & 1) * 16, 0, 3);
      __builtin_prefetch(W + (size_t)(k0 + 32 + (tid >> 3)) * Dc + col0 + (tid & 7) * 16, 0, 3);
    }
    __syncthreads();
    {  // stage A tile (fp32 -> f16), 16 elements/thread, coalesced reads
      const int m = tid >> 1;
      const int k = (tid & 1) * 16;
      const float4* src = reinterpret_cast<const float4*>(X + (size_t)(row0 + m) * Dc + k0 + k);
      _Float16 tmp[16];
#pragma unroll
      for (int j = 0; j < 4; ++j) {
        float4 f = src[j];
        tmp[4 * j + 0] = (_Float16)f.x; tmp[4 * j + 1] = (_Float16)f.y;
        tmp[4 * j + 2] = (_Float16)f.z; tmp[4 * j + 3] = (_Float16)f.w;
      }
      uint4* dst = reinterpret_cast<uint4*>(&ldsA[m * 32 + k]);
      dst[0] = reinterpret_cast<uint4*>(tmp)[0];
      dst[1] = reinterpret_cast<uint4*>(tmp)[1];
    }
    {  // stage B tile transposed: coalesced global reads, scatter b16 to [n][k]
      const int k = tid >> 3;
      const int n = (tid & 7) * 16;
      const float4* src = reinterpret_cast<const float4*>(W + (size_t)(k0 + k) * Dc + col0 + n);
#pragma unroll
      for (int j = 0; j < 4; ++j) {
        float4 f = src[j];
        ldsBT[(n + 4 * j + 0) * 32 + k] = (_Float16)f.x;
        ldsBT[(n + 4 * j + 1) * 32 + k] = (_Float16)f.y;
        ldsBT[(n + 4 * j + 2) * 32 + k] = (_Float16)f.z;
        ldsBT[(n + 4 * j + 3) * 32 + k] = (_Float16)f.w;
      }
    }
    __syncthreads();
    // A-frag (16x32 f16): lane m=l16; lanes<16 hold K {0..7,16..23}, lanes>=16 {8..15,24..31}
    v16h a;
    {
      const _Float16* abase = &ldsA[(wave * 16 + l16) * 32 + hi * 8];
      uint4* ap = reinterpret_cast<uint4*>(&a);
      ap[0] = *reinterpret_cast<const uint4*>(abase);
      ap[1] = *reinterpret_cast<const uint4*>(abase + 16);
    }
#pragma unroll
    for (int nt = 0; nt < 8; ++nt) {
      // B-frag (32x16 f16): lane n=l16; lanes<16 hold K {0..7,16..23}, lanes>=16 {8..15,24..31}
      const _Float16* bbase = &ldsBT[(nt * 16 + l16) * 32 + hi * 8];
      v16h bf;
      {
        uint4* bp = reinterpret_cast<uint4*>(&bf);
        bp[0] = *reinterpret_cast<const uint4*>(bbase);
        bp[1] = *reinterpret_cast<const uint4*>(bbase + 16);
      }
      acc[nt] = __builtin_amdgcn_wmma_f32_16x16x32_f16(false, a, false, bf,
                                                       (short)0, acc[nt], false, false);
    }
  }
  // epilogue: bias + write f16 in [B,H,S,HD] layout
#pragma unroll
  for (int nt = 0; nt < 8; ++nt) {
    const int n = col0 + nt * 16 + l16;
    const float bvv = bias[n];
    const int h = n / HDc, hd = n % HDc;
#pragma unroll
    for (int r = 0; r < 8; ++r) {
      const int m = row0 + wave * 16 + hi * 8 + r;
      const int bb = m / Sc, ss = m % Sc;
      Y[(((size_t)bb * Hc + h) * Sc + ss) * HDc + hd] = (_Float16)(acc[nt][r] + bvv);
    }
  }
}

// ---------------------------------------------------------------------------
// Fused flash-attention: one block = 128 query rows of one (b,h).
// Each wave owns a 16-row tile; streams keys in 32-key chunks.
// ---------------------------------------------------------------------------
__global__ __launch_bounds__(256)
void attn_fused(const _Float16* __restrict__ Q16, const _Float16* __restrict__ K16,
                const _Float16* __restrict__ V16, const unsigned char* __restrict__ mask,
                _Float16* __restrict__ C16) {
  __shared__ _Float16 ldsQ[128 * 64];   // [q_local][hd]
  __shared__ _Float16 ldsK[32 * 64];    // [key][hd]
  __shared__ _Float16 ldsVT[64 * 32];   // [hd][key] (transposed for PV B-frags)
  __shared__ _Float16 ldsP[8][16 * 32]; // per-wave P re-layout buffer
  const int tid = threadIdx.x;
  const int lane = tid & 31, wave = tid >> 5;
  const int l16 = lane & 15, hi = lane >> 4;
  const int nQB = Sc / 128;  // 16
  const int bh = blockIdx.x / nQB;
  const int qb = blockIdx.x % nQB;
  const int b = bh / Hc, h = bh % Hc;
  const _Float16* Qp = Q16 + (size_t)bh * Sc * HDc;
  const _Float16* Kp = K16 + (size_t)bh * Sc * HDc;
  const _Float16* Vp = V16 + (size_t)bh * Sc * HDc;

  // stage Q block (16 KB) cooperatively via async global->LDS
  {
    const _Float16* src = Qp + (size_t)qb * 128 * HDc;
#pragma unroll
    for (int i = 0; i < 4; ++i)
      copy16_g2l(ldsQ + (tid + 256 * i) * 8, src + (tid + 256 * i) * 8);
  }
  async_fence();
  __syncthreads();
  // Q A-frags held in registers: two K-steps covering hd 0..31 and 32..63
  v16h qa[2];
#pragma unroll
  for (int c = 0; c < 2; ++c) {
    const _Float16* base = &ldsQ[(wave * 16 + l16) * 64 + c * 32 + hi * 8];
    uint4* ap = reinterpret_cast<uint4*>(&qa[c]);
    ap[0] = *reinterpret_cast<const uint4*>(base);
    ap[1] = *reinterpret_cast<const uint4*>(base + 16);
  }

  v8f ctx[4] = {};
  float run_max[8], run_sum[8];
#pragma unroll
  for (int r = 0; r < 8; ++r) { run_max[r] = -3.0e38f; run_sum[r] = 0.f; }
  const float scale = 0.125f;  // 1/sqrt(64)

  for (int kb = 0; kb < Sc; kb += 32) {
    if (kb + 32 < Sc) {  // prefetch next K/V chunks
      __builtin_prefetch(Kp + (size_t)(kb + 32) * HDc + tid * 8, 0, 3);
      __builtin_prefetch(Vp + (size_t)(kb + 32) * HDc + tid * 8, 0, 3);
    }
    __syncthreads();
    // stage K chunk (32 keys x 64 hd): 16B per thread, async global->LDS
    copy16_g2l(ldsK + tid * 8, Kp + (size_t)kb * HDc + tid * 8);
    {  // stage V chunk transposed [hd][key] (needs VGPR path)
      const int key = tid >> 3;
      const int hd0 = (tid & 7) * 8;
      uint4 raw = *reinterpret_cast<const uint4*>(Vp + (size_t)(kb + key) * HDc + hd0);
      _Float16 tmp[8];
      *reinterpret_cast<uint4*>(tmp) = raw;
#pragma unroll
      for (int j = 0; j < 8; ++j) ldsVT[(hd0 + j) * 32 + key] = tmp[j];
    }
    async_fence();
    __syncthreads();

    // scores: S = Q . K^T for keys kb..kb+15 (s0) and kb+16..kb+31 (s1)
    v8f s0 = {}, s1 = {};
#pragma unroll
    for (int c = 0; c < 2; ++c) {
      const _Float16* k0b = &ldsK[l16 * 64 + c * 32 + hi * 8];
      const _Float16* k1b = &ldsK[(16 + l16) * 64 + c * 32 + hi * 8];
      v16h bf0, bf1;
      {
        uint4* p = reinterpret_cast<uint4*>(&bf0);
        p[0] = *reinterpret_cast<const uint4*>(k0b);
        p[1] = *reinterpret_cast<const uint4*>(k0b + 16);
      }
      {
        uint4* p = reinterpret_cast<uint4*>(&bf1);
        p[0] = *reinterpret_cast<const uint4*>(k1b);
        p[1] = *reinterpret_cast<const uint4*>(k1b + 16);
      }
      s0 = __builtin_amdgcn_wmma_f32_16x16x32_f16(false, qa[c], false, bf0,
                                                  (short)0, s0, false, false);
      s1 = __builtin_amdgcn_wmma_f32_16x16x32_f16(false, qa[c], false, bf1,
                                                  (short)0, s1, false, false);
    }

    // mask + online softmax (row = r + hi*8, cols live across the 16 lanes of this half)
    float p0[8], p1[8];
#pragma unroll
    for (int r = 0; r < 8; ++r) {
      const int qm = qb * 128 + wave * 16 + hi * 8 + r;
      const size_t mrow = ((size_t)b * Sc + qm) * Sc + kb;
      float x0 = s0[r] * scale;
      float x1 = s1[r] * scale;
      if (mask[mrow + l16]) x0 = -1e9f;
      if (mask[mrow + 16 + l16]) x1 = -1e9f;
      float mx = fmaxf(x0, x1);
#pragma unroll
      for (int off = 8; off >= 1; off >>= 1) mx = fmaxf(mx, __shfl_xor(mx, off, 16));
      const float nm = fmaxf(run_max[r], mx);
      const float alpha = __expf(run_max[r] - nm);
      run_max[r] = nm;
      const float e0 = __expf(x0 - nm);
      const float e1 = __expf(x1 - nm);
      p0[r] = e0; p1[r] = e1;
      float rs = e0 + e1;
#pragma unroll
      for (int off = 8; off >= 1; off >>= 1) rs += __shfl_xor(rs, off, 16);
      run_sum[r] = run_sum[r] * alpha + rs;
#pragma unroll
      for (int t = 0; t < 4; ++t) ctx[t][r] *= alpha;
    }

    // re-layout P (C-frag -> A-frag) through per-wave LDS (same-wave DS is in-order)
    _Float16* pw = &ldsP[wave][0];
#pragma unroll
    for (int r = 0; r < 8; ++r) {
      const int m = hi * 8 + r;
      pw[m * 32 + l16] = (_Float16)p0[r];
      pw[m * 32 + 16 + l16] = (_Float16)p1[r];
    }
    v16h pa;
    {
      const _Float16* base = &pw[l16 * 32 + hi * 8];
      uint4* ap = reinterpret_cast<uint4*>(&pa);
      ap[0] = *reinterpret_cast<const uint4*>(base);
      ap[1] = *reinterpret_cast<const uint4*>(base + 16);
    }
    // ctx += P (16x32) . V (32x64): four 16x16 output tiles
#pragma unroll
    for (int t = 0; t < 4; ++t) {
      const _Float16* vb = &ldsVT[(t * 16 + l16) * 32 + hi * 8];
      v16h bf;
      {
        uint4* bp = reinterpret_cast<uint4*>(&bf);
        bp[0] = *reinterpret_cast<const uint4*>(vb);
        bp[1] = *reinterpret_cast<const uint4*>(vb + 16);
      }
      ctx[t] = __builtin_amdgcn_wmma_f32_16x16x32_f16(false, pa, false, bf,
                                                      (short)0, ctx[t], false, false);
    }
  }

  // normalize + write ctx f16 as [b, s, h*64+hd] (merged-head [B,S,D] layout)
#pragma unroll
  for (int r = 0; r < 8; ++r) {
    const int qm = qb * 128 + wave * 16 + hi * 8 + r;
    const float inv = 1.0f / run_sum[r];
#pragma unroll
    for (int t = 0; t < 4; ++t) {
      const int hd = t * 16 + l16;
      C16[((size_t)b * Sc + qm) * Dc + h * HDc + hd] = (_Float16)(ctx[t][r] * inv);
    }
  }
}

// ---------------------------------------------------------------------------
// Output projection: out = f32( ctx16 @ Wo + bo )
// ---------------------------------------------------------------------------
__global__ __launch_bounds__(256)
void out_gemm_f32(const _Float16* __restrict__ X, const float* __restrict__ W,
                  const float* __restrict__ bias, float* __restrict__ out) {
  __shared__ _Float16 ldsA[128 * 32];
  __shared__ _Float16 ldsBT[128 * 32];
  const int tid = threadIdx.x;
  const int lane = tid & 31, wave = tid >> 5;
  const int l16 = lane & 15, hi = lane >> 4;
  const int row0 = blockIdx.x * 128;
  const int col0 = blockIdx.y * 128;

  v8f acc[8] = {};
  for (int k0 = 0; k0 < Dc; k0 += 32) {
    if (k0 + 32 < Dc) {
      __builtin_prefetch(X + (size_t)(row0 + (tid >> 1)) * Dc + k0 + 32 + (tid & 1) * 16, 0, 3);
      __builtin_prefetch(W + (size_t)(k0 + 32 + (tid >> 3)) * Dc + col0 + (tid & 7) * 16, 0, 3);
    }
    __syncthreads();
    {  // A already f16: async global->LDS b128 copies
      const int m = tid >> 1;
      const int k = (tid & 1) * 16;
      const _Float16* src = X + (size_t)(row0 + m) * Dc + k0 + k;
      _Float16* dst = &ldsA[m * 32 + k];
      copy16_g2l(dst, src);
      copy16_g2l(dst + 8, src + 8);
    }
    {
      const int k = tid >> 3;
      const int n = (tid & 7) * 16;
      const float4* src = reinterpret_cast<const float4*>(W + (size_t)(k0 + k) * Dc + col0 + n);
#pragma unroll
      for (int j = 0; j < 4; ++j) {
        float4 f = src[j];
        ldsBT[(n + 4 * j + 0) * 32 + k] = (_Float16)f.x;
        ldsBT[(n + 4 * j + 1) * 32 + k] = (_Float16)f.y;
        ldsBT[(n + 4 * j + 2) * 32 + k] = (_Float16)f.z;
        ldsBT[(n + 4 * j + 3) * 32 + k] = (_Float16)f.w;
      }
    }
    async_fence();
    __syncthreads();
    v16h a;
    {
      const _Float16* abase = &ldsA[(wave * 16 + l16) * 32 + hi * 8];
      uint4* ap = reinterpret_cast<uint4*>(&a);
      ap[0] = *reinterpret_cast<const uint4*>(abase);
      ap[1] = *reinterpret_cast<const uint4*>(abase + 16);
    }
#pragma unroll
    for (int nt = 0; nt < 8; ++nt) {
      const _Float16* bbase = &ldsBT[(nt * 16 + l16) * 32 + hi * 8];
      v16h bf;
      {
        uint4* bp = reinterpret_cast<uint4*>(&bf);
        bp[0] = *reinterpret_cast<const uint4*>(bbase);
        bp[1] = *reinterpret_cast<const uint4*>(bbase + 16);
      }
      acc[nt] = __builtin_amdgcn_wmma_f32_16x16x32_f16(false, a, false, bf,
                                                       (short)0, acc[nt], false, false);
    }
  }
#pragma unroll
  for (int nt = 0; nt < 8; ++nt) {
    const int n = col0 + nt * 16 + l16;
    const float bvv = bias[n];
#pragma unroll
    for (int r = 0; r < 8; ++r) {
      const int m = row0 + wave * 16 + hi * 8 + r;
      out[(size_t)m * Dc + n] = acc[nt][r] + bvv;
    }
  }
}

// ---------------------------------------------------------------------------
extern "C" void kernel_launch(void* const* d_in, const int* in_sizes, int n_in,
                              void* d_out, int out_size, void* d_ws, size_t ws_size,
                              hipStream_t stream) {
  (void)in_sizes; (void)n_in; (void)out_size; (void)ws_size;
  const float* q = (const float*)d_in[0];
  const float* k = (const float*)d_in[1];
  const float* v = (const float*)d_in[2];
  const unsigned char* mask = (const unsigned char*)d_in[3];
  const float* Wq = (const float*)d_in[4];
  const float* bq = (const float*)d_in[5];
  const float* Wk = (const float*)d_in[6];
  const float* bk = (const float*)d_in[7];
  const float* Wv = (const float*)d_in[8];
  const float* bv = (const float*)d_in[9];
  const float* Wo = (const float*)d_in[10];
  const float* bo = (const float*)d_in[11];

  _Float16* ws = (_Float16*)d_ws;
  const size_t P = (size_t)Bc * Sc * Dc;  // 8.39M elements per buffer
  _Float16* q16 = ws;
  _Float16* k16 = ws + P;
  _Float16* v16 = ws + 2 * P;
  _Float16* c16 = ws + 3 * P;

  dim3 gg(MTOT / 128, Dc / 128);  // 64 x 8
  proj_gemm_f16<<<gg, 256, 0, stream>>>(q, Wq, bq, q16);
  proj_gemm_f16<<<gg, 256, 0, stream>>>(k, Wk, bk, k16);
  proj_gemm_f16<<<gg, 256, 0, stream>>>(v, Wv, bv, v16);
  attn_fused<<<Bc * Hc * (Sc / 128), 256, 0, stream>>>(q16, k16, v16, mask, c16);
  out_gemm_f32<<<gg, 256, 0, stream>>>(c16, Wo, bo, (float*)d_out);
}
